// CA_SA_v4_12240656793949
// MI455X (gfx1250) — compile-verified, matrix operand order checked by reading, not run on hardware
//
#include <hip/hip_runtime.h>
#include <math.h>

// ---------------------------------------------------------------------------
// CDNA5 (gfx1250) wave32 WMMA types/helpers
// ---------------------------------------------------------------------------
typedef __attribute__((ext_vector_type(16))) __bf16 v16bf;
typedef __attribute__((ext_vector_type(8)))  __bf16 v8bf;
typedef __attribute__((ext_vector_type(4)))  __bf16 v4bf;
typedef __attribute__((ext_vector_type(8)))  float  v8f;

__device__ __forceinline__ v8f wmma_bf16(v16bf a, v16bf b, v8f c) {
    // D = A(16x32 bf16) x B(32x16 bf16) + C(16x16 f32)
    return __builtin_amdgcn_wmma_f32_16x16x32_bf16(
        /*neg_a=*/false, a, /*neg_b=*/false, b,
        /*c_mod=*/(short)0, c, /*reuse_a=*/false, /*reuse_b=*/false);
}

// Per-lane async 16B copy global -> LDS (ASYNCcnt-tracked, bypasses VGPRs).
// lds_off is the workgroup-relative LDS byte offset (low 32 bits of a generic
// pointer to a __shared__ object). Both addresses must be 16B aligned.
__device__ __forceinline__ void async_copy_b128(unsigned lds_off, const void* g) {
    asm volatile("global_load_async_to_lds_b128 %0, %1, off"
                 :: "v"(lds_off), "v"((unsigned long long)(uintptr_t)g)
                 : "memory");
}
__device__ __forceinline__ void wait_async0() {
    asm volatile("s_wait_asynccnt 0" ::: "memory");
}
__device__ __forceinline__ unsigned lds_off(const void* p) {
    return (unsigned)(uintptr_t)p;
}

// Load one 16x32 bf16 operand fragment from a row-major tile.
//   A operand: row index = M (lane%16), packed K = (h/8)*16 + (lane/16)*8 + h%8
//   B operand: pass B^T row-major (row index = N); identical packing.
// Requires: base 16B-aligned, ld (elements) a multiple of 8 -> b128 loads.
__device__ __forceinline__ v16bf load_frag(const __bf16* base, int ld) {
    const int lane = threadIdx.x & 31;
    const __bf16* p = base + (size_t)(lane & 15) * ld + ((lane >> 4) << 3);
    v8bf lo = *(const v8bf*)p;
    v8bf hi = *(const v8bf*)(p + 16);
    v16bf r;
#pragma unroll
    for (int i = 0; i < 8; ++i) { r[i] = lo[i]; r[i + 8] = hi[i]; }
    return r;
}

// ---------------------------------------------------------------------------
// Generic bf16 WMMA GEMM:  D[M,N] = W[M,K](f32) * X[K,N] + bias, per batch z.
//   B operand source:
//     Bconv != null : implicit im2col 3x3 SAME conv from NCHW f32 (K = convC*9)
//     else          : pixel-major bf16  Bpm[n][k]  (K multiple of 32; async LDS)
//   Epilogue (any subset): f32 [M][N], bf16 [M][N], bf16 transposed [N][M]
// Block = 128 threads (4 waves), 64x64 tile, K-step 32.
// ---------------------------------------------------------------------------
__global__ void __launch_bounds__(128)
casa_wmma_gemm(const float* __restrict__ A, const __bf16* __restrict__ Bpm,
               const float* __restrict__ Bconv, const float* __restrict__ bias,
               float* __restrict__ out_f32, __bf16* __restrict__ out_pm,
               __bf16* __restrict__ out_cm,
               int M, int N, int K, int convC, int convH, int convW, int relu) {
    __shared__ __align__(16) __bf16 As[64][40];
    __shared__ __align__(16) __bf16 Bs[64][40];

    const int tid  = threadIdx.x;
    const int wave = tid >> 5;
    const int lane = tid & 31;
    const int mb = blockIdx.y * 64;
    const int nb = blockIdx.x * 64;
    const size_t ob = (size_t)blockIdx.z * M * N;
    const __bf16* Bb = Bpm   ? Bpm   + (size_t)blockIdx.z * N * K     : nullptr;
    const float*  Cb = Bconv ? Bconv + (size_t)blockIdx.z * convC * N : nullptr;

    v8f acc[4] = {};

    for (int kb = 0; kb < K; kb += 32) {
        // Stage A tile: 64 rows x 32 (f32 -> bf16), vectorized b128 / b64
#pragma unroll
        for (int idx = tid; idx < 64 * 8; idx += 128) {
            int r = idx >> 3, c4 = (idx & 7) << 2;
            float4 v = *(const float4*)&A[(size_t)(mb + r) * K + kb + c4];
            v4bf o = {(__bf16)v.x, (__bf16)v.y, (__bf16)v.z, (__bf16)v.w};
            *(v4bf*)&As[r][c4] = o;
        }
        // Stage B^T tile: 64 n-rows x 32 k
        if (Cb) {
#pragma unroll 4
            for (int idx = tid; idx < 64 * 32; idx += 128) {
                int r = idx >> 5, c = idx & 31;
                int k = kb + c;
                int ch = k / 9, rem = k - ch * 9;
                int ky = rem / 3, kx = rem - ky * 3;
                int n = nb + r;
                int y = n / convW, x = n - y * convW;
                int yy = y + ky - 1, xx = x + kx - 1;
                float v = 0.f;
                if (yy >= 0 && yy < convH && xx >= 0 && xx < convW)
                    v = Cb[((size_t)ch * convH + yy) * convW + xx];
                Bs[r][c] = (__bf16)v;
            }
        } else {
            // async global->LDS, 16B per lane-op (ASYNCcnt)
#pragma unroll
            for (int idx = tid; idx < 64 * 4; idx += 128) {
                int r = idx >> 2, c8 = (idx & 3) << 3;
                async_copy_b128(lds_off(&Bs[r][c8]),
                                Bb + (size_t)(nb + r) * K + kb + c8);
            }
        }
        wait_async0();
        __syncthreads();

        v16bf af = load_frag(&As[wave * 16][0], 40);
#pragma unroll
        for (int t = 0; t < 4; ++t) {
            v16bf bf = load_frag(&Bs[t * 16][0], 40);
            acc[t] = wmma_bf16(af, bf, acc[t]);
        }
        __syncthreads();
    }

    // Epilogue: C layout -> element (M = r + 8*(lane/16), N = lane%16)
    const int g = lane >> 4, col = lane & 15;
    const int m0 = mb + wave * 16 + 8 * g;
#pragma unroll
    for (int t = 0; t < 4; ++t) {
        int n = nb + t * 16 + col;
#pragma unroll
        for (int r = 0; r < 8; ++r) {
            int m = m0 + r;
            float v = acc[t][r] + bias[m];
            if (relu) v = fmaxf(v, 0.f);
            if (out_f32) out_f32[ob + (size_t)m * N + n] = v;
            if (out_pm)  out_pm [ob + (size_t)n * M + m] = (__bf16)v;
            if (out_cm)  out_cm [ob + (size_t)m * N + n] = (__bf16)v;
        }
    }
}

// ---------------------------------------------------------------------------
// Flash attention (no scale, softmax over keys):
//   R[n][c] = sum_m softmax_m(Q[n,:].K[m,:]) * V[c][m]
// Q,K pixel-major bf16 [4096][256]; V channel-major bf16 [256][4096];
// R pixel-major bf16 [4096][256].
// Block = 128 thr (4 waves); each wave owns 16 query rows, full d=256 output
// accumulated in 16 WMMA C-tiles. 32-key steps, online softmax. K/V tiles are
// staged with GLOBAL_LOAD_ASYNC_TO_LDS_B128. Row stats live per-lane at index
// r for row (r + 8*(lane/16)) — exactly the WMMA C layout.
// ---------------------------------------------------------------------------
__global__ void __launch_bounds__(128)
casa_flash_attn(const __bf16* __restrict__ Q1, const __bf16* __restrict__ K1,
                const __bf16* __restrict__ V1, __bf16* __restrict__ R1,
                const __bf16* __restrict__ Q2, const __bf16* __restrict__ K2,
                const __bf16* __restrict__ V2, __bf16* __restrict__ R2) {
    const int NN = 4096, D = 256;
    __shared__ __align__(16) __bf16 Ks[32][264];   // 32 keys x 256 ch
    __shared__ __align__(16) __bf16 Vs[256][40];   // 256 ch x 32 keys (V^T rows = ch)
    __shared__ __align__(16) __bf16 Ps[4][16][40]; // per-wave P bounce 16x32

    const int tid = threadIdx.x, wave = tid >> 5, lane = tid & 31;
    const size_t off = (size_t)blockIdx.y * NN * D;
    const __bf16 *Q, *K, *V; __bf16* R;
    if (blockIdx.z == 0) { Q = Q1 + off; K = K1 + off; V = V1 + off; R = R1 + off; }
    else                 { Q = Q2 + off; K = K2 + off; V = V2 + off; R = R2 + off; }

    const int qb = blockIdx.x * 64 + wave * 16;

    // Q fragments resident in registers (8 chunks of 16x32)
    v16bf qf[8];
    {
        const __bf16* qbase = Q + (size_t)qb * D;
#pragma unroll
        for (int c = 0; c < 8; ++c) qf[c] = load_frag(qbase + c * 32, D);
    }

    v8f O[16] = {};
    float mrow[8], lrow[8];
#pragma unroll
    for (int r = 0; r < 8; ++r) { mrow[r] = -1e30f; lrow[r] = 0.f; }

    for (int kb = 0; kb < NN; kb += 32) {
        // async stage: K rows (32 x 512B) and V^T rows (256 x 64B)
#pragma unroll
        for (int idx = tid; idx < 32 * 32; idx += 128) {
            int r = idx >> 5, c8 = (idx & 31) << 3;
            async_copy_b128(lds_off(&Ks[r][c8]), K + (size_t)(kb + r) * D + c8);
        }
#pragma unroll
        for (int idx = tid; idx < 256 * 4; idx += 128) {
            int c = idx >> 2, m8 = (idx & 3) << 3;
            async_copy_b128(lds_off(&Vs[c][m8]), V + (size_t)c * NN + kb + m8);
        }
        wait_async0();
        __syncthreads();

        // S tiles for keys [kb,kb+16) and [kb+16,kb+32)
        v8f sa = {}, sb = {};
#pragma unroll
        for (int c = 0; c < 8; ++c) {
            v16bf ka  = load_frag(&Ks[0][c * 32], 264);
            v16bf kb2 = load_frag(&Ks[16][c * 32], 264);
            sa = wmma_bf16(qf[c], ka, sa);
            sb = wmma_bf16(qf[c], kb2, sb);
        }

        // online softmax update (row reductions within 16-lane halves)
        float scale[8], pa[8], pb[8];
#pragma unroll
        for (int r = 0; r < 8; ++r) {
            float v = fmaxf(sa[r], sb[r]);
            v = fmaxf(v, __shfl_xor(v, 1, 16));
            v = fmaxf(v, __shfl_xor(v, 2, 16));
            v = fmaxf(v, __shfl_xor(v, 4, 16));
            v = fmaxf(v, __shfl_xor(v, 8, 16));
            float mn = fmaxf(mrow[r], v);
            scale[r] = __expf(mrow[r] - mn);
            mrow[r] = mn;
            pa[r] = __expf(sa[r] - mn);
            pb[r] = __expf(sb[r] - mn);
            float s = pa[r] + pb[r];
            s += __shfl_xor(s, 1, 16);
            s += __shfl_xor(s, 2, 16);
            s += __shfl_xor(s, 4, 16);
            s += __shfl_xor(s, 8, 16);
            lrow[r] = lrow[r] * scale[r] + s;
        }
#pragma unroll
        for (int t = 0; t < 16; ++t)
#pragma unroll
            for (int r = 0; r < 8; ++r) O[t][r] *= scale[r];

        // P: C layout -> A layout through per-wave LDS bounce
        __bf16* pbuf = &Ps[wave][0][0];
        const int g = lane >> 4, col = lane & 15;
#pragma unroll
        for (int r = 0; r < 8; ++r) {
            pbuf[(size_t)(r + 8 * g) * 40 + col]      = (__bf16)pa[r];
            pbuf[(size_t)(r + 8 * g) * 40 + 16 + col] = (__bf16)pb[r];
        }
        asm volatile("s_wait_dscnt 0" ::: "memory"); // wave-local LDS RAW fence
        v16bf pf = load_frag(pbuf, 40);

        // O += P(16x32) x V^T(32x16) for all 16 channel tiles
#pragma unroll
        for (int t = 0; t < 16; ++t) {
            v16bf vf = load_frag(&Vs[t * 16][0], 40);
            O[t] = wmma_bf16(pf, vf, O[t]);
        }
        __syncthreads();
    }

    // normalize and store R pixel-major bf16
    const int g = lane >> 4, col = lane & 15;
#pragma unroll
    for (int r = 0; r < 8; ++r) {
        float inv = 1.0f / lrow[r];
        size_t n = (size_t)(qb + r + 8 * g);
#pragma unroll
        for (int t = 0; t < 16; ++t)
            R[n * D + t * 16 + col] = (__bf16)(O[t][r] * inv);
    }
}

// ---------------------------------------------------------------------------
// MVN stats per (b,c): mean and std = sqrt(unbiased var + 1e-5)
// ---------------------------------------------------------------------------
__global__ void __launch_bounds__(256)
casa_stats(const float* __restrict__ x, float* __restrict__ mean,
           float* __restrict__ stdv, int HW) {
    __shared__ float ss[256], ss2[256];
    const int bc = blockIdx.x;
    const float* p = x + (size_t)bc * HW;
    float s = 0.f, s2 = 0.f;
    for (int i = threadIdx.x; i < HW; i += 256) { float v = p[i]; s += v; s2 = fmaf(v, v, s2); }
    ss[threadIdx.x] = s; ss2[threadIdx.x] = s2;
    __syncthreads();
    for (int o = 128; o > 0; o >>= 1) {
        if (threadIdx.x < o) { ss[threadIdx.x] += ss[threadIdx.x + o]; ss2[threadIdx.x] += ss2[threadIdx.x + o]; }
        __syncthreads();
    }
    if (threadIdx.x == 0) {
        float m = ss[0] / (float)HW;
        float var = (ss2[0] - (float)HW * m * m) / (float)(HW - 1);
        mean[bc] = m;
        stdv[bc] = sqrtf(var + 1e-5f);
    }
}

// normalize F_content -> f32 channel-major + bf16 pixel-major
__global__ void __launch_bounds__(256)
casa_norm_content(const float* __restrict__ x, const float* __restrict__ mean,
                  const float* __restrict__ stdv, float* __restrict__ out_cm,
                  __bf16* __restrict__ out_pm) {
    size_t i = (size_t)blockIdx.x * 256 + threadIdx.x;
    int hw = (int)(i & 4095); size_t bc = i >> 12; int c = (int)(bc & 255); int b = (int)(bc >> 8);
    float v = (x[i] - mean[bc]) / stdv[bc];
    out_cm[i] = v;
    out_pm[(((size_t)b << 12) + hw) * 256 + c] = (__bf16)v;
}

// normalize Fs -> bf16 pixel-major (v) and (v*v)
__global__ void __launch_bounds__(256)
casa_norm_style(const float* __restrict__ x, const float* __restrict__ mean,
                const float* __restrict__ stdv, __bf16* __restrict__ o1,
                __bf16* __restrict__ o2) {
    size_t i = (size_t)blockIdx.x * 256 + threadIdx.x;
    int hw = (int)(i & 4095); size_t bc = i >> 12; int c = (int)(bc & 255); int b = (int)(bc >> 8);
    float v = (x[i] - mean[bc]) / stdv[bc];
    size_t pi = (((size_t)b << 12) + hw) * 256 + c;
    o1[pi] = (__bf16)v;
    o2[pi] = (__bf16)(v * v);
}

// r = o1 + o2 + fc
__global__ void __launch_bounds__(256)
casa_combine(const float* __restrict__ o1, const float* __restrict__ o2,
             const float* __restrict__ fc, float* __restrict__ r) {
    size_t i = (size_t)blockIdx.x * 256 + threadIdx.x;
    r[i] = o1[i] + o2[i] + fc[i];
}

// rn = mvn(r) * std_s + mean_s  -> bf16 pixel-major
__global__ void __launch_bounds__(256)
casa_final_mod(const float* __restrict__ r, const float* __restrict__ mr,
               const float* __restrict__ sr, const float* __restrict__ ms,
               const float* __restrict__ ss_, __bf16* __restrict__ out_pm) {
    size_t i = (size_t)blockIdx.x * 256 + threadIdx.x;
    int hw = (int)(i & 4095); size_t bc = i >> 12; int c = (int)(bc & 255); int b = (int)(bc >> 8);
    float v = (r[i] - mr[bc]) / sr[bc] * ss_[bc] + ms[bc];
    out_pm[(((size_t)b << 12) + hw) * 256 + c] = (__bf16)v;
}

// ---------------------------------------------------------------------------
extern "C" void kernel_launch(void* const* d_in, const int* in_sizes, int n_in,
                              void* d_out, int out_size, void* d_ws, size_t ws_size,
                              hipStream_t stream) {
    (void)in_sizes; (void)n_in; (void)out_size; (void)ws_size;
    const int B = 2, C = 256, CLIP = 512, H = 64, W = 64, HW = 64 * 64, N = HW;

    const float* F_clip   = (const float*)d_in[0];
    const float* F_cont   = (const float*)d_in[1];
    const float* fs_w1    = (const float*)d_in[2];
    const float* fs_b1    = (const float*)d_in[3];
    const float* fs_w2    = (const float*)d_in[4];
    const float* fs_b2    = (const float*)d_in[5];
    const float* out_w    = (const float*)d_in[6];
    const float* out_b    = (const float*)d_in[7];
    const float* a1_fw = (const float*)d_in[8],  *a1_fb = (const float*)d_in[9];
    const float* a1_gw = (const float*)d_in[10], *a1_gb = (const float*)d_in[11];
    const float* a1_hw = (const float*)d_in[12], *a1_hb = (const float*)d_in[13];
    const float* a1_ow = (const float*)d_in[14], *a1_ob = (const float*)d_in[15];
    const float* a2_fw = (const float*)d_in[16], *a2_fb = (const float*)d_in[17];
    const float* a2_gw = (const float*)d_in[18], *a2_gb = (const float*)d_in[19];
    const float* a2_hw = (const float*)d_in[20], *a2_hb = (const float*)d_in[21];
    const float* a2_ow = (const float*)d_in[22], *a2_ob = (const float*)d_in[23];

    // workspace carve
    char* w = (char*)d_ws;
    auto alloc = [&](size_t bytes) { char* p = w; w += (bytes + 255) & ~size_t(255); return (void*)p; };
    const size_t F32SZ = (size_t)B * C * HW * sizeof(float);
    const size_t BFSZ  = (size_t)B * C * HW * sizeof(__bf16);
    float*  T1     = (float*)alloc(F32SZ);   // relu(conv1)
    float*  Fs_cm  = (float*)alloc(F32SZ);
    float*  Fc_cm  = (float*)alloc(F32SZ);
    __bf16* Fc_pm  = (__bf16*)alloc(BFSZ);
    __bf16* Fs_pm  = (__bf16*)alloc(BFSZ);
    __bf16* Fs2_pm = (__bf16*)alloc(BFSZ);
    __bf16* F1 = (__bf16*)alloc(BFSZ); __bf16* G1 = (__bf16*)alloc(BFSZ); __bf16* H1 = (__bf16*)alloc(BFSZ);
    __bf16* F2 = (__bf16*)alloc(BFSZ); __bf16* G2 = (__bf16*)alloc(BFSZ); __bf16* H2 = (__bf16*)alloc(BFSZ);
    __bf16* R1 = (__bf16*)alloc(BFSZ); __bf16* R2 = (__bf16*)alloc(BFSZ);
    float*  O1 = (float*)alloc(F32SZ); float* O2 = (float*)alloc(F32SZ);
    float*  rr = (float*)alloc(F32SZ);
    __bf16* rn = (__bf16*)alloc(BFSZ);
    float* mean_c = (float*)alloc(B * C * 4); float* std_c = (float*)alloc(B * C * 4);
    float* mean_s = (float*)alloc(B * C * 4); float* std_s = (float*)alloc(B * C * 4);
    float* mean_r = (float*)alloc(B * C * 4); float* std_r = (float*)alloc(B * C * 4);

    const dim3 gblk(128);
    const dim3 ggrid(N / 64, C / 64, B);
    const dim3 eblk(256);
    const int  egrid = (B * C * HW) / 256;

    // 1) Fs = conv3x3(relu(conv3x3(F_clip_s)))
    casa_wmma_gemm<<<ggrid, gblk, 0, stream>>>(fs_w1, nullptr, F_clip, fs_b1,
        T1, nullptr, nullptr, C, N, CLIP * 9, CLIP, H, W, /*relu=*/1);
    casa_wmma_gemm<<<ggrid, gblk, 0, stream>>>(fs_w2, nullptr, T1, fs_b2,
        Fs_cm, nullptr, nullptr, C, N, C * 9, C, H, W, /*relu=*/0);

    // 2) MVN stats + normalized/pixel-major packs
    casa_stats<<<B * C, eblk, 0, stream>>>(F_cont, mean_c, std_c, HW);
    casa_stats<<<B * C, eblk, 0, stream>>>(Fs_cm, mean_s, std_s, HW);
    casa_norm_content<<<egrid, eblk, 0, stream>>>(F_cont, mean_c, std_c, Fc_cm, Fc_pm);
    casa_norm_style<<<egrid, eblk, 0, stream>>>(Fs_cm, mean_s, std_s, Fs_pm, Fs2_pm);

    // 3) 1x1 projections (Q/K pixel-major, V channel-major)
    casa_wmma_gemm<<<ggrid, gblk, 0, stream>>>(a1_fw, Fc_pm, nullptr, a1_fb,
        nullptr, F1, nullptr, C, N, C, 0, 0, 0, 0);
    casa_wmma_gemm<<<ggrid, gblk, 0, stream>>>(a1_gw, Fs_pm, nullptr, a1_gb,
        nullptr, G1, nullptr, C, N, C, 0, 0, 0, 0);
    casa_wmma_gemm<<<ggrid, gblk, 0, stream>>>(a1_hw, Fs_pm, nullptr, a1_hb,
        nullptr, nullptr, H1, C, N, C, 0, 0, 0, 0);
    casa_wmma_gemm<<<ggrid, gblk, 0, stream>>>(a2_fw, Fc_pm, nullptr, a2_fb,
        nullptr, F2, nullptr, C, N, C, 0, 0, 0, 0);
    casa_wmma_gemm<<<ggrid, gblk, 0, stream>>>(a2_gw, Fs2_pm, nullptr, a2_gb,
        nullptr, G2, nullptr, C, N, C, 0, 0, 0, 0);
    casa_wmma_gemm<<<ggrid, gblk, 0, stream>>>(a2_hw, Fs2_pm, nullptr, a2_hb,
        nullptr, nullptr, H2, C, N, C, 0, 0, 0, 0);

    // 4) fused flash attention (both attns, both batches)
    casa_flash_attn<<<dim3(N / 64, B, 2), gblk, 0, stream>>>(F1, G1, H1, R1,
                                                             F2, G2, H2, R2);

    // 5) out-projections + residual
    casa_wmma_gemm<<<ggrid, gblk, 0, stream>>>(a1_ow, R1, nullptr, a1_ob,
        O1, nullptr, nullptr, C, N, C, 0, 0, 0, 0);
    casa_wmma_gemm<<<ggrid, gblk, 0, stream>>>(a2_ow, R2, nullptr, a2_ob,
        O2, nullptr, nullptr, C, N, C, 0, 0, 0, 0);
    casa_combine<<<egrid, eblk, 0, stream>>>(O1, O2, Fc_cm, rr);

    // 6) r = mvn(r)*std_s + mean_s ; out = conv1x1(r, out_w, out_b)
    casa_stats<<<B * C, eblk, 0, stream>>>(rr, mean_r, std_r, HW);
    casa_final_mod<<<egrid, eblk, 0, stream>>>(rr, mean_r, std_r, mean_s, std_s, rn);
    casa_wmma_gemm<<<ggrid, gblk, 0, stream>>>(out_w, rn, nullptr, out_b,
        (float*)d_out, nullptr, nullptr, C, N, C, 0, 0, 0, 0);
}
